// BurgersDifferentiator_88510686036690
// MI455X (gfx1250) — matrix operand order repaired and checked by reading, not compiled, not run on hardware
//
#include <hip/hip_runtime.h>
#include <hip/hip_bf16.h>
#include <math.h>

typedef __attribute__((ext_vector_type(16))) __bf16 v16bf;
typedef __attribute__((ext_vector_type(8)))  float  v8f;
typedef int v4i __attribute__((vector_size(16)));

#define NEG_SLOPE 0.2f
#define GLOBAL_AS __attribute__((address_space(1)))
#define LDS_AS    __attribute__((address_space(3)))

__device__ __forceinline__ void atomicMaxF(float* addr, float val) {
    if (val >= 0.0f) atomicMax((int*)addr, __float_as_int(val));
    else             atomicMin((unsigned int*)addr, __float_as_uint(val));
}

__global__ void k_fill(float* __restrict__ p, float v, int n) {
    int t = blockIdx.x * blockDim.x + threadIdx.x;
    if (t < n) p[t] = v;
}

// ---- Edge pass 1: degree, static sums, least-squares accumulators ----
__global__ void k_edge_geom(const float* __restrict__ fs, const int* __restrict__ ei,
                            float* __restrict__ deg, float* __restrict__ sstat,
                            float* __restrict__ gA, float* __restrict__ gB,
                            float* __restrict__ Mm, float* __restrict__ R, int nE) {
    int e = blockIdx.x * blockDim.x + threadIdx.x;
    if (e >= nE) return;
    int s = ei[e], d = ei[nE + e];
    const float* fss = fs + (size_t)s * 5;
    const float* fsd = fs + (size_t)d * 5;
    float s0 = fss[0], s1 = fss[1], s2 = fss[2];
    float d0 = s0 - fsd[0];
    float d1 = s1 - fsd[1];
    float df0 = fss[3] - fsd[3];
    float df1 = fss[4] - fsd[4];
    atomicAdd(deg + d, 1.0f);
    atomicAdd(sstat + 3 * d + 0, s0);
    atomicAdd(sstat + 3 * d + 1, s1);
    atomicAdd(sstat + 3 * d + 2, s2);
    atomicAdd(gA + 3 * d + 0, d0 * d0);
    atomicAdd(gA + 3 * d + 1, d0 * d1);
    atomicAdd(gA + 3 * d + 2, d1 * d1);
    atomicAdd(gB + 4 * d + 0, d0 * df0);
    atomicAdd(gB + 4 * d + 1, d0 * df1);
    atomicAdd(gB + 4 * d + 2, d1 * df0);
    atomicAdd(gB + 4 * d + 3, d1 * df1);
    float P[5] = {d0, d1, 0.5f * d0 * d0, d0 * d1, 0.5f * d1 * d1};
    float* Md = Mm + (size_t)d * 15;
    int idx = 0;
    #pragma unroll
    for (int i = 0; i < 5; ++i)
        #pragma unroll
        for (int j = i; j < 5; ++j)
            atomicAdd(Md + (idx++), P[i] * P[j]);
    float* Rd = R + (size_t)d * 10;
    #pragma unroll
    for (int i = 0; i < 5; ++i) {
        atomicAdd(Rd + 2 * i + 0, P[i] * df0);
        atomicAdd(Rd + 2 * i + 1, P[i] * df1);
    }
}

// ---- Node pass 1: h1 = relu(static@Ws1 + mean_static@Wn1 + b1) ----
__global__ void k_node_l1(const float* __restrict__ fs, const float* __restrict__ sstat,
                          const float* __restrict__ deg,
                          const float* __restrict__ Ws1, const float* __restrict__ Wn1,
                          const float* __restrict__ b1, float* __restrict__ h1, int nN) {
    int t = blockIdx.x * blockDim.x + threadIdx.x;
    if (t >= nN * 64) return;
    int n = t >> 6, f = t & 63;
    float inv = 1.0f / fmaxf(deg[n], 1.0f);
    const float* st = fs + (size_t)n * 5;
    const float* ss = sstat + (size_t)n * 3;
    float acc = b1[f];
    #pragma unroll
    for (int j = 0; j < 3; ++j)
        acc += st[j] * Ws1[j * 64 + f] + (ss[j] * inv) * Wn1[j * 64 + f];
    h1[(size_t)n * 64 + f] = fmaxf(acc, 0.0f);
}

// ---- Edge pass 2: scatter-add h1[src] into sum_h1[dst] (wave per edge) ----
__global__ void k_edge_scatter(const float* __restrict__ h1, const int* __restrict__ ei,
                               float* __restrict__ sumh1, int nE) {
    long long t = (long long)blockIdx.x * blockDim.x + threadIdx.x;
    long long e = t >> 5;
    if (e >= nE) return;
    int lane = (int)(t & 31);
    int s = ei[e], d = ei[(long long)nE + e];
    const float* hs = h1 + (size_t)s * 64;
    float* sd = sumh1 + (size_t)d * 64;
    atomicAdd(sd + lane,      hs[lane]);
    atomicAdd(sd + lane + 32, hs[lane + 32]);
}

// ---- Convert W_self2/W_nbr2 to bf16 column-major: Wt[m][n*64+k] = bf16(W[k*64+n]) ----
__global__ void k_cvt_w(const float* __restrict__ Ws2, const float* __restrict__ Wn2,
                        __bf16* __restrict__ Wt) {
    int t = blockIdx.x * blockDim.x + threadIdx.x;
    if (t >= 4096) return;
    int k = t >> 6, n = t & 63;
    Wt[n * 64 + k]        = (__bf16)Ws2[k * 64 + n];
    Wt[4096 + n * 64 + k] = (__bf16)Wn2[k * 64 + n];
}

// ---- Node pass 2 (WMMA): h2 = relu(h1@Ws2 + mean_h1@Wn2 + b2); feature stats ----
// B fragments come from LDS-staged bf16 column-major weights (16 KB / block).
__global__ void k_node_l2_wmma(const float* __restrict__ h1, const float* __restrict__ sumh1,
                               const float* __restrict__ deg,
                               const __bf16* __restrict__ Wt,   // [2][64*64] col-major bf16
                               const float* __restrict__ b2, float* __restrict__ h2,
                               float* __restrict__ feat, int ntiles) {
    __shared__ __bf16 ldsW[8192];   // 16 KB: [0,4096)=self, [4096,8192)=nbr

    // Stage weights into LDS (all 256 threads participate).
#if __has_builtin(__builtin_amdgcn_global_load_async_to_lds_b128)
    #pragma unroll
    for (int i = 0; i < 4; ++i) {
        int idx = ((int)threadIdx.x + i * 256) * 8;   // 8 bf16 = 16 B per b128
        __builtin_amdgcn_global_load_async_to_lds_b128(
            (GLOBAL_AS v4i*)(Wt + idx),
            (LDS_AS v4i*)(&ldsW[idx]), 0, 0);
    }
#if __has_builtin(__builtin_amdgcn_s_wait_asynccnt)
    __builtin_amdgcn_s_wait_asynccnt(0);
#else
    asm volatile("s_wait_asynccnt 0x0" ::: "memory");
#endif
#else
    // Fallback: plain global load + ds_store
    #pragma unroll
    for (int i = 0; i < 4; ++i) {
        int idx = ((int)threadIdx.x + i * 256) * 8;
        *(v16bf*)(&ldsW[idx]) = *(const v16bf*)(Wt + idx);  // 32 B (2 halves)
    }
#endif
    __syncthreads();

    int lane = threadIdx.x & 31;
    int wave = (blockIdx.x * blockDim.x + threadIdx.x) >> 5;
    if (wave >= ntiles) return;   // wave-uniform: EXEC stays all-ones below

    int row = wave * 16 + (lane & 15);
    int kbase = (lane < 16) ? 0 : 16;
    float invdeg = 1.0f / fmaxf(deg[row], 1.0f);
    v16bf aS[2], aN[2];
    #pragma unroll
    for (int ks = 0; ks < 2; ++ks) {
        const float* hp = h1    + (size_t)row * 64 + ks * 32 + kbase;
        const float* sp = sumh1 + (size_t)row * 64 + ks * 32 + kbase;
        #pragma unroll
        for (int e = 0; e < 8; ++e) {
            aS[ks][e]     = (__bf16)hp[e];
            aS[ks][e + 8] = (__bf16)hp[e + 8];
            aN[ks][e]     = (__bf16)(sp[e] * invdeg);
            aN[ks][e + 8] = (__bf16)(sp[e + 8] * invdeg);
        }
    }
    int kb16 = (lane < 16) ? 0 : 16;
    int rbase = wave * 16 + ((lane < 16) ? 0 : 8);
    #pragma unroll
    for (int nt = 0; nt < 4; ++nt) {
        int col = nt * 16 + (lane & 15);
        // 16 contiguous bf16 (32 B, 32 B-aligned) per fragment from LDS
        int fb = col * 64 + kb16;
        v16bf bS0 = *(const v16bf*)(&ldsW[fb]);
        v16bf bS1 = *(const v16bf*)(&ldsW[fb + 32]);
        v16bf bN0 = *(const v16bf*)(&ldsW[4096 + fb]);
        v16bf bN1 = *(const v16bf*)(&ldsW[4096 + fb + 32]);
        v8f c = {};
        c = __builtin_amdgcn_wmma_f32_16x16x32_bf16(false, aS[0], false, bS0, (short)0, c, false, false);
        c = __builtin_amdgcn_wmma_f32_16x16x32_bf16(false, aS[1], false, bS1, (short)0, c, false, false);
        c = __builtin_amdgcn_wmma_f32_16x16x32_bf16(false, aN[0], false, bN0, (short)0, c, false, false);
        c = __builtin_amdgcn_wmma_f32_16x16x32_bf16(false, aN[1], false, bN1, (short)0, c, false, false);
        float bias = b2[col];
        float ps = 0.0f, ps2 = 0.0f;
        #pragma unroll
        for (int v = 0; v < 8; ++v) {
            float val = fmaxf(c[v] + bias, 0.0f);
            h2[(size_t)(rbase + v) * 64 + col] = val;
            ps += val; ps2 += val * val;
        }
        ps  += __shfl_xor(ps, 16, 32);
        ps2 += __shfl_xor(ps2, 16, 32);
        if (lane < 16) {
            atomicAdd(feat + col, ps);
            atomicAdd(feat + 64 + col, ps2);
        }
    }
}

// scalar tail for rows not covered by full 16-row tiles
__global__ void k_node_l2_tail(const float* __restrict__ h1, const float* __restrict__ sumh1,
                               const float* __restrict__ deg,
                               const float* __restrict__ Ws2, const float* __restrict__ Wn2,
                               const float* __restrict__ b2, float* __restrict__ h2,
                               float* __restrict__ feat, int startRow, int nRows) {
    int t = blockIdx.x * blockDim.x + threadIdx.x;
    if (t >= nRows * 64) return;
    int n = startRow + (t >> 6), f = t & 63;
    float inv = 1.0f / fmaxf(deg[n], 1.0f);
    float acc = b2[f];
    for (int k = 0; k < 64; ++k)
        acc += h1[(size_t)n * 64 + k] * Ws2[k * 64 + f] + sumh1[(size_t)n * 64 + k] * inv * Wn2[k * 64 + f];
    float val = fmaxf(acc, 0.0f);
    h2[(size_t)n * 64 + f] = val;
    atomicAdd(feat + f, val);
    atomicAdd(feat + 64 + f, val * val);
}

__global__ void k_stats(const float* __restrict__ feat, float* __restrict__ stats, int nN) {
    int f = threadIdx.x;
    if (f < 64) {
        float mu  = feat[f] / (float)nN;
        float var = feat[64 + f] / (float)nN - mu * mu;
        stats[f] = mu;
        stats[64 + f] = rsqrtf(var + 1e-5f);
    }
}

// ---- Per-node 2x2 and 5x5 solves -> physics_mask ----
__global__ void k_node_solve(const float* __restrict__ fs, const float* __restrict__ gA,
                             const float* __restrict__ gB, const float* __restrict__ Mm,
                             const float* __restrict__ R, float* __restrict__ pm, int nN) {
    int n = blockIdx.x * blockDim.x + threadIdx.x;
    if (n >= nN) return;
    float a00 = gA[3 * n + 0] + 1e-6f;
    float a01 = gA[3 * n + 1];
    float a11 = gA[3 * n + 2] + 1e-6f;
    float id = 1.0f / (a00 * a11 - a01 * a01);
    float b00 = gB[4 * n + 0], b01 = gB[4 * n + 1], b10 = gB[4 * n + 2], b11 = gB[4 * n + 3];
    float g00 = ( a11 * b00 - a01 * b10) * id;
    float g01 = ( a11 * b01 - a01 * b11) * id;
    float g10 = (-a01 * b00 + a00 * b10) * id;
    float g11 = (-a01 * b01 + a00 * b11) * id;
    float v0 = fs[(size_t)n * 5 + 3], v1 = fs[(size_t)n * 5 + 4];
    float adv0 = -(v0 * g00 + v1 * g10);
    float adv1 = -(v0 * g01 + v1 * g11);
    float A[5][5], X[5][2];
    const float* Mn = Mm + (size_t)n * 15;
    int idx = 0;
    #pragma unroll
    for (int i = 0; i < 5; ++i)
        #pragma unroll
        for (int j = i; j < 5; ++j) { float v = Mn[idx++]; A[i][j] = v; A[j][i] = v; }
    #pragma unroll
    for (int i = 0; i < 5; ++i) A[i][i] += 1e-6f;
    const float* Rn = R + (size_t)n * 10;
    #pragma unroll
    for (int i = 0; i < 5; ++i) { X[i][0] = Rn[2 * i]; X[i][1] = Rn[2 * i + 1]; }
    #pragma unroll
    for (int k = 0; k < 5; ++k) {
        float piv = 1.0f / A[k][k];
        #pragma unroll
        for (int i = k + 1; i < 5; ++i) {
            float fct = A[i][k] * piv;
            #pragma unroll
            for (int j = k; j < 5; ++j) A[i][j] -= fct * A[k][j];
            X[i][0] -= fct * X[k][0];
            X[i][1] -= fct * X[k][1];
        }
    }
    #pragma unroll
    for (int k = 4; k >= 0; --k) {
        float x0 = X[k][0], x1 = X[k][1];
        #pragma unroll
        for (int j = k + 1; j < 5; ++j) { x0 -= A[k][j] * X[j][0]; x1 -= A[k][j] * X[j][1]; }
        float ip = 1.0f / A[k][k];
        X[k][0] = x0 * ip; X[k][1] = x1 * ip;
    }
    pm[4 * n + 0] = adv0;
    pm[4 * n + 1] = adv1;
    pm[4 * n + 2] = X[2][0] + X[4][0];
    pm[4 * n + 3] = X[2][1] + X[4][1];
}

// ---- FiLM modulation + GAT projection: wave per node ----
__global__ void k_node_gat(const float* __restrict__ h2, const float* __restrict__ stats,
                           const float* __restrict__ pm,
                           const float* __restrict__ Wg, const float* __restrict__ bg,
                           const float* __restrict__ Wb, const float* __restrict__ bb,
                           const float* __restrict__ Wgat, const float* __restrict__ a_s,
                           const float* __restrict__ a_d,
                           float* __restrict__ z, float* __restrict__ zs, float* __restrict__ zd,
                           int nN) {
    int t = blockIdx.x * blockDim.x + threadIdx.x;
    int n = t >> 5, lane = t & 31;
    if (n >= nN) return;
    float p0 = pm[4 * n + 0], p1 = pm[4 * n + 1], p2 = pm[4 * n + 2], p3 = pm[4 * n + 3];
    float z0 = 0.0f, z1 = 0.0f;
    #pragma unroll
    for (int q = 0; q < 2; ++q) {
        int f = lane + q * 32;
        float mu = stats[f], rs = stats[64 + f];
        float xn = (h2[(size_t)n * 64 + f] - mu) * rs;
        float gamma = p0 * Wg[f] + p1 * Wg[64 + f] + p2 * Wg[128 + f] + p3 * Wg[192 + f] + bg[f];
        float beta  = p0 * Wb[f] + p1 * Wb[64 + f] + p2 * Wb[128 + f] + p3 * Wb[192 + f] + bb[f];
        float xm = xn * (1.0f + gamma) + beta;
        z0 += xm * Wgat[f * 2 + 0];
        z1 += xm * Wgat[f * 2 + 1];
    }
    #pragma unroll
    for (int off = 16; off >= 1; off >>= 1) {
        z0 += __shfl_xor(z0, off, 32);
        z1 += __shfl_xor(z1, off, 32);
    }
    if (lane == 0) {
        z[2 * n + 0] = z0; z[2 * n + 1] = z1;
        zs[n] = z0 * a_s[0] + z1 * a_s[1];
        zd[n] = z0 * a_d[0] + z1 * a_d[1];
    }
}

__global__ void k_edge_max(const int* __restrict__ ei, const float* __restrict__ zs,
                           const float* __restrict__ zd, float* __restrict__ mx, int nE) {
    int e = blockIdx.x * blockDim.x + threadIdx.x;
    if (e >= nE) return;
    int s = ei[e], d = ei[nE + e];
    float L = zs[s] + zd[d];
    L = (L > 0.0f) ? L : L * NEG_SLOPE;
    atomicMaxF(mx + d, L);
}

__global__ void k_edge_sum(const int* __restrict__ ei, const float* __restrict__ zs,
                           const float* __restrict__ zd, const float* __restrict__ mx,
                           const float* __restrict__ z, float* __restrict__ denom,
                           float* __restrict__ oacc, int nE) {
    int e = blockIdx.x * blockDim.x + threadIdx.x;
    if (e >= nE) return;
    int s = ei[e], d = ei[nE + e];
    float L = zs[s] + zd[d];
    L = (L > 0.0f) ? L : L * NEG_SLOPE;
    float w = expf(L - mx[d]);
    atomicAdd(denom + d, w);
    atomicAdd(oacc + 2 * d + 0, w * z[2 * s + 0]);
    atomicAdd(oacc + 2 * d + 1, w * z[2 * s + 1]);
}

__global__ void k_out(const float* __restrict__ oacc, const float* __restrict__ denom,
                      const float* __restrict__ bgat, float* __restrict__ out, int nN) {
    int t = blockIdx.x * blockDim.x + threadIdx.x;
    if (t >= 2 * nN) return;
    int n = t >> 1, j = t & 1;
    out[t] = oacc[t] / (denom[n] + 1e-16f) + bgat[j];
}

extern "C" void kernel_launch(void* const* d_in, const int* in_sizes, int n_in,
                              void* d_out, int out_size, void* d_ws, size_t ws_size,
                              hipStream_t stream) {
    const float* fs   = (const float*)d_in[0];
    const float* Ws1  = (const float*)d_in[1];
    const float* Wn1  = (const float*)d_in[2];
    const float* b1   = (const float*)d_in[3];
    const float* Ws2  = (const float*)d_in[4];
    const float* Wn2  = (const float*)d_in[5];
    const float* b2   = (const float*)d_in[6];
    const float* Wg   = (const float*)d_in[7];
    const float* bg   = (const float*)d_in[8];
    const float* Wb   = (const float*)d_in[9];
    const float* bb   = (const float*)d_in[10];
    const float* Wgat = (const float*)d_in[11];
    const float* a_s  = (const float*)d_in[12];
    const float* a_d  = (const float*)d_in[13];
    const float* bgat = (const float*)d_in[14];
    const int*   ei   = (const int*)d_in[15];

    const int nN = in_sizes[0] / 5;
    const int nE = in_sizes[15] / 2;
    const size_t N = (size_t)nN;

    float* W     = (float*)d_ws;
    float* deg   = W;                 // N     (zero)
    float* sstat = W + N;             // 3N    (zero)
    float* gA    = W + 4 * N;         // 3N    (zero)
    float* gB    = W + 7 * N;         // 4N    (zero)
    float* Mm    = W + 11 * N;        // 15N   (zero)
    float* R     = W + 26 * N;        // 10N   (zero)
    float* sumh1 = W + 36 * N;        // 64N   (zero)
    float* denom = W + 100 * N;       // N     (zero)
    float* oacc  = W + 101 * N;       // 2N    (zero)
    float* feat  = W + 103 * N;       // 128   (zero)
    float* mx    = W + 103 * N + 128; // N     (fill -inf)
    float* h1    = W + 104 * N + 128; // 64N
    float* h2    = W + 168 * N + 128; // 64N
    float* pm    = W + 232 * N + 128; // 4N
    float* z     = W + 236 * N + 128; // 2N
    float* zs    = W + 238 * N + 128; // N
    float* zd    = W + 239 * N + 128; // N
    float* stats = W + 240 * N + 128; // 128
    __bf16* Wt   = (__bf16*)(W + 240 * N + 256); // 8192 bf16 = 16 KB (4096 floats)

    const int T = 256;
    (void)hipMemsetAsync(d_ws, 0, (103 * N + 128) * sizeof(float), stream);
    k_fill<<<(nN + T - 1) / T, T, 0, stream>>>(mx, -INFINITY, nN);

    k_edge_geom<<<(nE + T - 1) / T, T, 0, stream>>>(fs, ei, deg, sstat, gA, gB, Mm, R, nE);
    k_node_l1<<<(nN * 64 + T - 1) / T, T, 0, stream>>>(fs, sstat, deg, Ws1, Wn1, b1, h1, nN);

    long long scatterThreads = (long long)nE * 32;
    k_edge_scatter<<<(unsigned)((scatterThreads + T - 1) / T), T, 0, stream>>>(h1, ei, sumh1, nE);

    k_cvt_w<<<(4096 + T - 1) / T, T, 0, stream>>>(Ws2, Wn2, Wt);

    int ntiles = nN / 16;
    if (ntiles > 0) {
        long long wmmaThreads = (long long)ntiles * 32;
        k_node_l2_wmma<<<(unsigned)((wmmaThreads + T - 1) / T), T, 0, stream>>>(
            h1, sumh1, deg, Wt, b2, h2, feat, ntiles);
    }
    int remRows = nN - ntiles * 16;
    if (remRows > 0) {
        k_node_l2_tail<<<(remRows * 64 + T - 1) / T, T, 0, stream>>>(
            h1, sumh1, deg, Ws2, Wn2, b2, h2, feat, ntiles * 16, remRows);
    }
    k_stats<<<1, 64, 0, stream>>>(feat, stats, nN);

    k_node_solve<<<(nN + T - 1) / T, T, 0, stream>>>(fs, gA, gB, Mm, R, pm, nN);
    k_node_gat<<<((long long)nN * 32 + T - 1) / T, T, 0, stream>>>(
        h2, stats, pm, Wg, bg, Wb, bb, Wgat, a_s, a_d, z, zs, zd, nN);

    k_edge_max<<<(nE + T - 1) / T, T, 0, stream>>>(ei, zs, zd, mx, nE);
    k_edge_sum<<<(nE + T - 1) / T, T, 0, stream>>>(ei, zs, zd, mx, z, denom, oacc, nE);
    k_out<<<(2 * nN + T - 1) / T, T, 0, stream>>>(oacc, denom, bgat, (float*)d_out, nN);
}
